// RuleEmbedding_11003706213183
// MI455X (gfx1250) — matrix operand order repaired and checked by reading, not compiled, not run on hardware
//
#include <hip/hip_runtime.h>
#include <hip/hip_bf16.h>
#include <math.h>

typedef __attribute__((ext_vector_type(16))) _Float16 v16h;
typedef __attribute__((ext_vector_type(8)))  float    v8f;

#define R_ 32
#define D_ 128
#define B_ 32
#define N_ 32768            // R^3 rules
#define WAVES_PER_BLOCK 8
#define RULES_PER_WAVE 16

// workspace layout (in floats)
#define WS_QH 0                        // 128*32   qh_t[d][b] (includes b1)
#define WS_SC 4096                     // 32*32768 scores[b][n]
#define WS_BH (4096 + B_ * N_)         // 32768*128 bh[n][d]

// ---------------------------------------------------------------------------
// K1: qh_t[d][b] = (query @ w1[:128]) + b1, stored transposed for conflict-free
// per-lane (lane==b) broadcast reads in K2.
// ---------------------------------------------------------------------------
__global__ void k1_qh(const float* __restrict__ query,
                      const float* __restrict__ w1,
                      const float* __restrict__ b1,
                      float* __restrict__ qh_t) {
    __shared__ float sq[B_ * D_];
    const int tid = threadIdx.x;               // 128 threads
    for (int idx = tid; idx < B_ * D_; idx += blockDim.x) sq[idx] = query[idx];
    __syncthreads();
    const int d = tid;
    const float bias = b1[d];
    for (int b = 0; b < B_; ++b) {
        float acc = bias;
#pragma unroll 8
        for (int k = 0; k < D_; ++k) acc += sq[b * D_ + k] * w1[k * D_ + d];
        qh_t[d * B_ + b] = acc;
    }
}

// ---------------------------------------------------------------------------
// K2: per wave: 16 rules. body = rel[i]*rel[j]*rel[k] built as f16 WMMA A
// fragments directly in registers; b_h = body @ w1b via v_wmma_f32_16x16x32_f16
// (4 K-chunks x 8 N-tiles = 32 WMMAs). Then score[b][n] = relu-dot with w2.
// ---------------------------------------------------------------------------
__global__ void __launch_bounds__(256)
k2_scores(const float* __restrict__ rel,
          const float* __restrict__ w1,
          const float* __restrict__ w2,
          const float* __restrict__ qh_t,
          float* __restrict__ scores,
          float* __restrict__ bh) {
    // w1b pre-swizzled into per-fragment B layout (32 frags x 32 lanes x 16 halves)
    __shared__ __align__(16) _Float16 sW1b[32 * 512];   // 32 KB
    __shared__ float sRel[R_ * D_];                     // 16 KB
    __shared__ float sW2[D_];

    const int tid = threadIdx.x;

    for (int idx = tid; idx < R_ * D_; idx += blockDim.x) sRel[idx] = rel[idx];
    if (tid < D_) sW2[tid] = w2[tid];

    // Stage w1b (w1 rows 128..255) as B fragments.
    // Dense 16-bit B (32x16) layout: VGPR v<4, lanes 0-15: K=2v,2v+1 (v>=4: K=8+2v,9+2v) col=lane;
    // lanes 16-31: same K + 8, col=lane-16.
    for (int idx = tid; idx < 32 * 32 * 8; idx += blockDim.x) {
        const int f    = idx >> 8;          // fragment = kc*8 + dt
        const int lane = (idx >> 3) & 31;
        const int dw   = idx & 7;           // dword within lane's 16 halves
        const int kc = f >> 3, dt = f & 7;
        const int col = dt * 16 + (lane & 15);
        const int kb  = (lane & 16) ? 8 : 0;
        const int kk  = kc * 32 + kb + dw * 2 + ((dw >= 4) ? 8 : 0);
        sW1b[f * 512 + lane * 16 + dw * 2 + 0] = (_Float16)w1[(D_ + kk + 0) * D_ + col];
        sW1b[f * 512 + lane * 16 + dw * 2 + 1] = (_Float16)w1[(D_ + kk + 1) * D_ + col];
    }
    __syncthreads();

    const int wave     = tid >> 5;
    const int lane     = tid & 31;
    const int ruleBase = (blockIdx.x * WAVES_PER_BLOCK + wave) * RULES_PER_WAVE;

    // 16-bit A (16x32) layout: lane<16 row=lane K-base 0; lane>=16 row=lane-16 K-base 8.
    const int row  = lane & 15;
    const int rule = ruleBase + row;
    const int i0 = rule >> 10, i1 = (rule >> 5) & 31, i2 = rule & 31;
    const int kbaseA = (lane & 16) ? 8 : 0;

    v8f acc[8] = {};
#pragma unroll
    for (int kc = 0; kc < 4; ++kc) {
        v16h a;
#pragma unroll
        for (int h = 0; h < 16; ++h) {
            const int K = kc * 32 + kbaseA + h + ((h >= 8) ? 8 : 0);
            const float body = sRel[i0 * D_ + K] * sRel[i1 * D_ + K] * sRel[i2 * D_ + K];
            a[h] = (_Float16)body;
        }
#pragma unroll
        for (int dt = 0; dt < 8; ++dt) {
            const v16h bfr =
                *reinterpret_cast<const v16h*>(&sW1b[(kc * 8 + dt) * 512 + lane * 16]);
            acc[dt] = __builtin_amdgcn_wmma_f32_16x16x32_f16(
                false, a, false, bfr, (short)0, acc[dt], false, false);
        }
    }

    // C layout: VGPR r, lanes 0-15: M=r, N=lane; lanes 16-31: M=r+8, N=lane-16.
    // Spill the 16x128 b_h tile to global (L2-resident) for the cross-lane score phase.
    const int mo  = (lane & 16) ? 8 : 0;
    const int col = lane & 15;
#pragma unroll
    for (int dt = 0; dt < 8; ++dt) {
#pragma unroll
        for (int r = 0; r < 8; ++r) {
            bh[(ruleBase + mo + r) * D_ + dt * 16 + col] = acc[dt][r];
        }
    }
    asm volatile("s_wait_storecnt 0" ::: "memory");

    // Score phase: lane == batch index. sc[n] = sum_d relu(qh+bh)*w2
    float sc[16];
#pragma unroll
    for (int n = 0; n < 16; ++n) sc[n] = 0.f;
    for (int d = 0; d < D_; ++d) {
        const float qv = qh_t[d * B_ + lane];   // coalesced across lanes
        const float wv = sW2[d];
#pragma unroll
        for (int n = 0; n < 16; ++n) {
            const float t = qv + bh[(ruleBase + n) * D_ + d];  // broadcast, L2/L0 hit
            sc[n] += fmaxf(t, 0.f) * wv;
        }
    }
#pragma unroll
    for (int n = 0; n < 16; ++n)
        scores[lane * N_ + ruleBase + n] = sc[n];
}

// ---------------------------------------------------------------------------
// K3: one block per batch b. Softmax over N via max + exp; attention marginals
// over each hop index (LDS ds_add_f32 atomics); subgoals = (marg @ rel)/Z.
// ---------------------------------------------------------------------------
__global__ void k3_softmax_out(const float* __restrict__ scores,
                               const float* __restrict__ rel,
                               float* __restrict__ out) {
    const int b   = blockIdx.x;
    const int tid = threadIdx.x;    // 256 threads
    __shared__ float red[256];
    __shared__ float marg[96];      // [3][32] attention marginals
    __shared__ float sMax, sZ;

    const float* srow = scores + b * N_;

    // pass 1: max
    float m = -3.402823466e38f;
    for (int n = tid; n < N_; n += 256) m = fmaxf(m, srow[n]);
    red[tid] = m; __syncthreads();
    for (int s = 128; s > 0; s >>= 1) {
        if (tid < s) red[tid] = fmaxf(red[tid], red[tid + s]);
        __syncthreads();
    }
    if (tid == 0) sMax = red[0];
    if (tid < 96) marg[tid] = 0.f;
    __syncthreads();

    // pass 2: exp-sum + marginals (n = i*1024 + j*32 + k)
    float psum = 0.f;
    const float mx = sMax;
    for (int n = tid; n < N_; n += 256) {
        const float e = __expf(srow[n] - mx);
        psum += e;
        atomicAdd(&marg[n >> 10], e);
        atomicAdd(&marg[32 + ((n >> 5) & 31)], e);
        atomicAdd(&marg[64 + (n & 31)], e);
    }
    red[tid] = psum; __syncthreads();
    for (int s = 128; s > 0; s >>= 1) {
        if (tid < s) red[tid] += red[tid + s];
        __syncthreads();
    }
    if (tid == 0) sZ = red[0];
    __syncthreads();

    // subgoals[b,h,:] = sum_i marg[h][i] * rel[i,:] / Z
    const float invZ = 1.f / sZ;
    if (tid < D_) {
        const int d = tid;
#pragma unroll
        for (int h = 0; h < 3; ++h) {
            float a = 0.f;
#pragma unroll
            for (int i = 0; i < R_; ++i) a += marg[h * 32 + i] * rel[i * D_ + d];
            out[b * 3 * D_ + h * D_ + d] = a * invZ;
        }
    }
    // masks: all true -> 1.0f, appended after subgoals
    if (tid < 3) out[B_ * 3 * D_ + b * 3 + tid] = 1.0f;
}

extern "C" void kernel_launch(void* const* d_in, const int* in_sizes, int n_in,
                              void* d_out, int out_size, void* d_ws, size_t ws_size,
                              hipStream_t stream) {
    const float* query = (const float*)d_in[0];   // (32,128)
    const float* rel   = (const float*)d_in[1];   // (32,128)
    const float* w1    = (const float*)d_in[2];   // (256,128)
    const float* b1    = (const float*)d_in[3];   // (128,)
    const float* w2    = (const float*)d_in[4];   // (128,1)
    // d_in[5] = b2: uniform score shift -> softmax-invariant -> unused.
    (void)in_sizes; (void)n_in; (void)out_size; (void)ws_size;

    float* out    = (float*)d_out;
    float* wsf    = (float*)d_ws;
    float* qh_t   = wsf + WS_QH;
    float* scores = wsf + WS_SC;
    float* bh     = wsf + WS_BH;

    k1_qh<<<1, 128, 0, stream>>>(query, w1, b1, qh_t);
    k2_scores<<<N_ / (WAVES_PER_BLOCK * RULES_PER_WAVE), 256, 0, stream>>>(
        rel, w1, w2, qh_t, scores, bh);
    k3_softmax_out<<<B_, 256, 0, stream>>>(scores, rel, out);
}
